// Transformer_3315714752822
// MI455X (gfx1250) — matrix operand order repaired and checked by reading, not compiled, not run on hardware
//
#include <hip/hip_runtime.h>
#include <hip/hip_bf16.h>

// ---------------------------------------------------------------------------
// Transformer encoder-decoder forward for MI455X (gfx1250, wave32, WMMA).
// Dense math: v_wmma_f32_16x16x32_bf16.  Staging: global_load_async_to_lds
// (ASYNCcnt) with double-buffered LDS.
// ---------------------------------------------------------------------------

typedef __attribute__((ext_vector_type(16))) __bf16 v16bf;
typedef __attribute__((ext_vector_type(8)))  float  v8f;

union Frag {
    v16bf v;
    uint4 q[2];
    unsigned short h[16];
};

__device__ __forceinline__ unsigned short f2bf(float f) {
    union { float f; unsigned u; } x; x.f = f;
    unsigned r = x.u + 0x7FFFu + ((x.u >> 16) & 1u);   // round-to-nearest-even
    return (unsigned short)(r >> 16);
}

__device__ __forceinline__ float rmax16(float v) {
#pragma unroll
    for (int m = 8; m >= 1; m >>= 1) v = fmaxf(v, __shfl_xor(v, m, 16));
    return v;
}
__device__ __forceinline__ float rsum16(float v) {
#pragma unroll
    for (int m = 8; m >= 1; m >>= 1) v += __shfl_xor(v, m, 16);
    return v;
}

// 16B global -> LDS async copy (per-lane addresses), tracked by ASYNCcnt.
__device__ __forceinline__ void async_cp16(void* lds, const void* g) {
    unsigned la = (unsigned)(unsigned long long)lds;   // LDS offset = low 32b
    asm volatile("global_load_async_to_lds_b128 %0, %1, off"
                 :: "v"(la), "v"(g) : "memory");
}
__device__ __forceinline__ void wait_async0() {
    asm volatile("s_wait_asynccnt 0x0" ::: "memory");
}

// --------------------------- fp32 -> bf16 convert ---------------------------
__global__ __launch_bounds__(256)
void cvt_f32_bf16(const float* __restrict__ s, unsigned short* __restrict__ d,
                  size_t n) {
    size_t i = (size_t)blockIdx.x * blockDim.x + threadIdx.x;
    size_t stride = (size_t)gridDim.x * blockDim.x;
    for (; i < n; i += stride) d[i] = f2bf(s[i]);
}

// ------------------------------- GEMM (WMMA) --------------------------------
// C[M,N] = A[M,K] @ W[N,K]^T + bias[N], optional ReLU.
// Block: 256 thr = 8 waves; tile 128(M) x 128(N); wave tile 32x64 (2x4 WMMA).
// Double-buffered LDS filled by global_load_async_to_lds_b128.
__global__ __launch_bounds__(256)
void gemm_bf16_wmma(const unsigned short* __restrict__ A,
                    const unsigned short* __restrict__ W,
                    const float* __restrict__ bias,
                    float* __restrict__ C,
                    int M, int N, int K, int relu) {
    __shared__ unsigned short As[2][128][40];   // k-tile 32, pad 40 (80B rows)
    __shared__ unsigned short Bs[2][128][40];

    const int tid  = threadIdx.x;
    const int lane = tid & 31;
    const int hf   = lane >> 4;              // lane half (0: lanes 0-15)
    const int wid  = tid >> 5;               // 8 waves: 4(M) x 2(N)
    const int m0w  = (wid & 3) * 32;
    const int n0w  = (wid >> 2) * 64;
    const int m0   = blockIdx.y * 128;
    const int n0   = blockIdx.x * 128;

    v8f acc[2][4];
    const v8f vz = {0.f,0.f,0.f,0.f,0.f,0.f,0.f,0.f};
#pragma unroll
    for (int i = 0; i < 2; ++i)
#pragma unroll
        for (int j = 0; j < 4; ++j) acc[i][j] = vz;

    // cooperative staging coords: 2 threads per row, 32B (2x16B) each
    const int sr = tid >> 1, sh = tid & 1;
    const unsigned short* Ap = A + (size_t)(m0 + sr) * K + sh * 16;
    const unsigned short* Wp = W + (size_t)(n0 + sr) * K + sh * 16;

    auto stage = [&](int bsel, int k0) {
        async_cp16(&As[bsel][sr][sh * 16],     Ap + k0);
        async_cp16(&As[bsel][sr][sh * 16 + 8], Ap + k0 + 8);
        async_cp16(&Bs[bsel][sr][sh * 16],     Wp + k0);
        async_cp16(&Bs[bsel][sr][sh * 16 + 8], Wp + k0 + 8);
    };

    const int nk = K >> 5;
    stage(0, 0);
    wait_async0();
    __syncthreads();

    int buf = 0;
    for (int t = 0; t < nk; ++t) {
        if (t + 1 < nk) stage(buf ^ 1, (t + 1) * 32);   // overlap next tile

        Frag fa[2], fb[4];
#pragma unroll
        for (int i = 0; i < 2; ++i) {      // A frags (16-bit A layout)
            int r = m0w + i * 16 + (lane & 15);
            fa[i].q[0] = *(const uint4*)&As[buf][r][hf * 8];      // k 0-7/8-15
            fa[i].q[1] = *(const uint4*)&As[buf][r][16 + hf * 8]; // k 16-23/24-31
        }
#pragma unroll
        for (int j = 0; j < 4; ++j) {      // B frags: lane = col, k by half
            int r = n0w + j * 16 + (lane & 15);
            fb[j].q[0] = *(const uint4*)&Bs[buf][r][hf * 8];      // k 0-7/8-15
            fb[j].q[1] = *(const uint4*)&Bs[buf][r][16 + hf * 8]; // k 16-23/24-31
        }
#pragma unroll
        for (int i = 0; i < 2; ++i)
#pragma unroll
            for (int j = 0; j < 4; ++j)
                acc[i][j] = __builtin_amdgcn_wmma_f32_16x16x32_bf16(
                    false, fa[i].v, false, fb[j].v, (short)0, acc[i][j],
                    false, false);

        wait_async0();          // next tile landed (this wave's issues)
        __syncthreads();        // all waves done reading buf / writing buf^1
        buf ^= 1;
    }

    // epilogue: bias (+ReLU); C layout: VGPR e -> M = e + 8*half, lane&15 -> N
#pragma unroll
    for (int j = 0; j < 4; ++j) {
        int gn = n0 + n0w + j * 16 + (lane & 15);
        float bj = bias[gn];
#pragma unroll
        for (int i = 0; i < 2; ++i) {
#pragma unroll
            for (int e = 0; e < 8; ++e) {
                int gm = m0 + m0w + i * 16 + e + (hf ? 8 : 0);
                float v = acc[i][j][e] + bj;
                if (relu) v = fmaxf(v, 0.f);
                C[(size_t)gm * N + gn] = v;
            }
        }
    }
}

// --------------------------- flash attention (WMMA) -------------------------
// Q,K,V: bf16 [B, L, H*64]; O: fp32 [B, Lq, H*64].  16 queries per wave,
// 32 keys per iteration. Scores: 4 WMMAs (dh=64 = 2 k-chunks of 32);
// online softmax with 16-lane shuffle reductions; P transposed through LDS
// into A layout; V staged in LDS (async) as B matrix; 4 AV WMMAs.
__global__ __launch_bounds__(256)
void flash_attn(const unsigned short* __restrict__ Q,
                const unsigned short* __restrict__ K,
                const unsigned short* __restrict__ V,
                float* __restrict__ O,
                int Lq, int Lk, int causal) {
    __shared__ unsigned short Pl[8][16][40];   // per-wave P tile (16x32)
    __shared__ unsigned short Vl[8][32][80];   // per-wave V tile (32x64, pad 80)

    const int lane = threadIdx.x & 31;
    const int hf   = lane >> 4;
    const int w    = threadIdx.x >> 5;
    const int q0   = blockIdx.x * 128 + w * 16;
    const int b    = blockIdx.y >> 4;
    const int h    = blockIdx.y & 15;
    const float SCALE = 0.125f;                // 1/sqrt(64)

    const v8f vz = {0.f,0.f,0.f,0.f,0.f,0.f,0.f,0.f};
    v8f o0 = vz, o1 = vz, o2 = vz, o3 = vz;
    float mrow[8], lrow[8];
#pragma unroll
    for (int e = 0; e < 8; ++e) { mrow[e] = -1e30f; lrow[e] = 0.f; }

    // preload Q A-frags for both dh chunks (16-bit A layout)
    Frag fq0, fq1;
    {
        int qr = q0 + (lane & 15);
        const unsigned short* qp = Q + ((size_t)b * Lq + qr) * 1024 + h * 64;
        fq0.q[0] = *(const uint4*)(qp + hf * 8);
        fq0.q[1] = *(const uint4*)(qp + 16 + hf * 8);
        fq1.q[0] = *(const uint4*)(qp + 32 + hf * 8);
        fq1.q[1] = *(const uint4*)(qp + 48 + hf * 8);
    }

    int kend = causal ? (((q0 + 16) + 31) & ~31) : Lk;
    if (kend > Lk) kend = Lk;

    for (int kb = 0; kb < kend; kb += 32) {
        // stage V tile async: each lane copies one key row (64 bf16 = 128B)
        {
            const unsigned short* vg =
                V + ((size_t)b * Lk + kb + lane) * 1024 + h * 64;
#pragma unroll
            for (int i = 0; i < 8; ++i)
                async_cp16(&Vl[w][lane][i * 8], vg + i * 8);
        }

        // scores: S[16q x 32k] as two 16x16 tiles, k-dim = dh = 64
        v8f s0 = vz, s1 = vz;
#pragma unroll
        for (int t = 0; t < 2; ++t) {
            int key = kb + t * 16 + (lane & 15);
            const unsigned short* kp = K + ((size_t)b * Lk + key) * 1024 + h * 64;
            Frag fk0, fk1;                      // B frags, dh chunks 0 / 1
            fk0.q[0] = *(const uint4*)(kp + hf * 8);
            fk0.q[1] = *(const uint4*)(kp + 16 + hf * 8);
            fk1.q[0] = *(const uint4*)(kp + 32 + hf * 8);
            fk1.q[1] = *(const uint4*)(kp + 48 + hf * 8);
            v8f st = (t == 0) ? s0 : s1;
            st = __builtin_amdgcn_wmma_f32_16x16x32_bf16(false, fq0.v, false,
                                                         fk0.v, (short)0, st,
                                                         false, false);
            st = __builtin_amdgcn_wmma_f32_16x16x32_bf16(false, fq1.v, false,
                                                         fk1.v, (short)0, st,
                                                         false, false);
            if (t == 0) s0 = st; else s1 = st;
        }

        // online softmax per query row (row e+8*hf, cols across 16 lanes)
#pragma unroll
        for (int e = 0; e < 8; ++e) {
            int qr = q0 + e + (hf ? 8 : 0);
            float sv0 = s0[e] * SCALE, sv1 = s1[e] * SCALE;
            if (causal) {
                int ke = kb + (lane & 15);
                if (ke > qr)      sv0 = -1e30f;
                if (ke + 16 > qr) sv1 = -1e30f;
            }
            float rm   = fmaxf(rmax16(sv0), rmax16(sv1));
            float mn   = fmaxf(mrow[e], rm);
            float corr = __expf(mrow[e] - mn);
            float p0   = __expf(sv0 - mn);
            float p1   = __expf(sv1 - mn);
            lrow[e] = lrow[e] * corr + rsum16(p0) + rsum16(p1);
            mrow[e] = mn;
            o0[e] *= corr; o1[e] *= corr; o2[e] *= corr; o3[e] *= corr;
            int pr = e + (hf ? 8 : 0);
            Pl[w][pr][lane & 15]        = f2bf(p0);
            Pl[w][pr][16 + (lane & 15)] = f2bf(p1);
        }

        // per-wave LDS RAW: async V tile + ds P tile must land before reads
        asm volatile("s_wait_asynccnt 0x0\n\ts_wait_dscnt 0x0" ::: "memory");

        // P as A-matrix (16x32 over keys)
        Frag fp;
        {
            int r = lane & 15;
            fp.q[0] = *(const uint4*)&Pl[w][r][hf * 8];
            fp.q[1] = *(const uint4*)&Pl[w][r][16 + hf * 8];
        }
        // V as B-matrix per dh tile of 16: lane = dh col, keys by half
#pragma unroll
        for (int d = 0; d < 4; ++d) {
            Frag fv;
            int col = d * 16 + (lane & 15);
#pragma unroll
            for (int kk = 0; kk < 16; ++kk)
                fv.h[kk] = Vl[w][hf * 8 + kk + (kk >= 8 ? 8 : 0)][col];
            v8f od = (d == 0) ? o0 : (d == 1) ? o1 : (d == 2) ? o2 : o3;
            od = __builtin_amdgcn_wmma_f32_16x16x32_bf16(false, fp.v, false,
                                                         fv.v, (short)0, od,
                                                         false, false);
            if (d == 0) o0 = od; else if (d == 1) o1 = od;
            else if (d == 2) o2 = od; else o3 = od;
        }
    }

    // normalize + store
#pragma unroll
    for (int e = 0; e < 8; ++e) {
        int qr = q0 + e + (hf ? 8 : 0);
        float inv = 1.f / lrow[e];
        float* op = O + ((size_t)b * Lq + qr) * 1024 + h * 64 + (lane & 15);
        op[0]  = o0[e] * inv;
        op[16] = o1[e] * inv;
        op[32] = o2[e] * inv;
        op[48] = o3[e] * inv;
    }
}

// -------------------------- fused residual + LayerNorm ----------------------
__global__ __launch_bounds__(256)
void add_ln_k(const float* __restrict__ x, const float* __restrict__ res,
              const float* __restrict__ g, const float* __restrict__ bta,
              float* __restrict__ out) {
    __shared__ float red[256];
    const int row = blockIdx.x, t = threadIdx.x;
    const float* xr = x   + (size_t)row * 1024;
    const float* rr = res + (size_t)row * 1024;
    float v[4]; float s = 0.f;
#pragma unroll
    for (int i = 0; i < 4; ++i) { v[i] = xr[t + 256 * i] + rr[t + 256 * i]; s += v[i]; }
    red[t] = s; __syncthreads();
    for (int st = 128; st > 0; st >>= 1) { if (t < st) red[t] += red[t + st]; __syncthreads(); }
    float mean = red[0] * (1.f / 1024.f); __syncthreads();
    float vs = 0.f;
#pragma unroll
    for (int i = 0; i < 4; ++i) { float d = v[i] - mean; vs += d * d; }
    red[t] = vs; __syncthreads();
    for (int st = 128; st > 0; st >>= 1) { if (t < st) red[t] += red[t + st]; __syncthreads(); }
    float rstd = rsqrtf(red[0] * (1.f / 1024.f) + 1e-5f);
#pragma unroll
    for (int i = 0; i < 4; ++i) {
        int c = t + 256 * i;
        out[(size_t)row * 1024 + c] = (v[i] - mean) * rstd * g[c] + bta[c];
    }
}

// ------------------------------ orchestration -------------------------------
extern "C" void kernel_launch(void* const* d_in, const int* in_sizes, int n_in,
                              void* d_out, int out_size, void* d_ws, size_t ws_size,
                              hipStream_t stream) {
    (void)in_sizes; (void)n_in; (void)out_size; (void)ws_size;
    const int Bsz = 4, L = 512, Dm = 1024, Hh = 16, FFd = 4096;
    const int M = Bsz * L;                           // 2048 tokens
    const size_t MD = (size_t)M * Dm, MF = (size_t)M * FFd;

    auto f = [&](int i) { return (const float*)d_in[i]; };

    // ---- workspace carve (~140 MB) ----
    char* wsp = (char*)d_ws; size_t off = 0;
    auto alloc = [&](size_t bytes) -> void* {
        void* p = wsp + off; off += (bytes + 255) & ~(size_t)255; return p;
    };
    unsigned short* wbuf = (unsigned short*)alloc((size_t)FFd * Dm * 2); // weight stage
    unsigned short* xb   = (unsigned short*)alloc(MD * 2);
    unsigned short* kvb  = (unsigned short*)alloc(MD * 2);
    unsigned short* qb   = (unsigned short*)alloc(MD * 2);
    unsigned short* kb2  = (unsigned short*)alloc(MD * 2);
    unsigned short* vb   = (unsigned short*)alloc(MD * 2);
    unsigned short* ob   = (unsigned short*)alloc(MD * 2);
    unsigned short* hb   = (unsigned short*)alloc(MF * 2);
    float* xe  = (float*)alloc(MD * 4);
    float* xd  = (float*)alloc(MD * 4);
    float* qf  = (float*)alloc(MD * 4);
    float* kf  = (float*)alloc(MD * 4);
    float* vf  = (float*)alloc(MD * 4);
    float* of  = (float*)alloc(MD * 4);
    float* t0  = (float*)alloc(MD * 4);
    float* hid = (float*)alloc(MF * 4);

    // ---- param leaves (jax tree: dict keys sorted alphabetically) ----
    struct AttnP { const float *Wk,*Wo,*Wq,*Wv,*bk,*bo,*bq,*bv; };
    struct EncL  { const float *W1,*W2,*b1,*b2,*n1b,*n1g,*n2b,*n2g; AttnP at; };
    struct DecL  { const float *W1,*W2,*b1,*b2,*n1b,*n1g,*n2b,*n2g,*n3b,*n3g;
                   AttnP cross, self; };
    int ix = 5;
    auto getAttn = [&]() { AttnP a;
        a.Wk=f(ix++); a.Wo=f(ix++); a.Wq=f(ix++); a.Wv=f(ix++);
        a.bk=f(ix++); a.bo=f(ix++); a.bq=f(ix++); a.bv=f(ix++); return a; };
    EncL enc[4]; DecL dec[4];
    for (int l = 0; l < 4; ++l) {   // W1,W2,attn,b1,b2,n1b,n1g,n2b,n2g
        enc[l].W1=f(ix++); enc[l].W2=f(ix++); enc[l].at=getAttn();
        enc[l].b1=f(ix++); enc[l].b2=f(ix++);
        enc[l].n1b=f(ix++); enc[l].n1g=f(ix++); enc[l].n2b=f(ix++); enc[l].n2g=f(ix++);
    }
    for (int l = 0; l < 4; ++l) {   // W1,W2,b1,b2,cross,n1b..n3g,self
        dec[l].W1=f(ix++); dec[l].W2=f(ix++); dec[l].b1=f(ix++); dec[l].b2=f(ix++);
        dec[l].cross=getAttn();
        dec[l].n1b=f(ix++); dec[l].n1g=f(ix++); dec[l].n2b=f(ix++); dec[l].n2g=f(ix++);
        dec[l].n3b=f(ix++); dec[l].n3g=f(ix++);
        dec[l].self=getAttn();
    }

    // ---- launch helpers ----
    auto cvt = [&](const float* s, unsigned short* d, size_t n) {
        cvt_f32_bf16<<<dim3(1024), dim3(256), 0, stream>>>(s, d, n);
    };
    auto gemm = [&](const unsigned short* A, const unsigned short* W,
                    const float* bias, float* C, int m, int n, int k, int relu) {
        gemm_bf16_wmma<<<dim3(n / 128, m / 128), dim3(256), 0, stream>>>(
            A, W, bias, C, m, n, k, relu);
    };
    auto lnadd = [&](const float* x, const float* res, const float* g,
                     const float* b, float* o) {
        add_ln_k<<<dim3(M), dim3(256), 0, stream>>>(x, res, g, b, o);
    };
    auto do_mha = [&](const float* xq, const float* xkv, const AttnP& a,
                      int causal, float* outp) {
        cvt(xq, xb, MD);
        cvt(a.Wq, wbuf, (size_t)Dm * Dm); gemm(xb,  wbuf, a.bq, qf, M, Dm, Dm, 0);
        cvt(xkv, kvb, MD);
        cvt(a.Wk, wbuf, (size_t)Dm * Dm); gemm(kvb, wbuf, a.bk, kf, M, Dm, Dm, 0);
        cvt(a.Wv, wbuf, (size_t)Dm * Dm); gemm(kvb, wbuf, a.bv, vf, M, Dm, Dm, 0);
        cvt(qf, qb, MD); cvt(kf, kb2, MD); cvt(vf, vb, MD);
        flash_attn<<<dim3(L / 128, Bsz * Hh), dim3(256), 0, stream>>>(
            qb, kb2, vb, of, L, L, causal);
        cvt(of, ob, MD);
        cvt(a.Wo, wbuf, (size_t)Dm * Dm); gemm(ob, wbuf, a.bo, outp, M, Dm, Dm, 0);
    };
    auto do_ffn = [&](const float* xcur, const float* W1, const float* b1,
                      const float* W2, const float* b2, float* outp) {
        cvt(xcur, xb, MD);
        cvt(W1, wbuf, (size_t)FFd * Dm); gemm(xb, wbuf, b1, hid, M, FFd, Dm, 1);
        cvt(hid, hb, MF);
        cvt(W2, wbuf, (size_t)Dm * FFd); gemm(hb, wbuf, b2, outp, M, Dm, FFd, 0);
    };

    // ---- encoder ----
    hipMemcpyAsync(xe, d_in[0], MD * 4, hipMemcpyDeviceToDevice, stream);
    for (int l = 0; l < 4; ++l) {
        do_mha(xe, xe, enc[l].at, 0, t0);
        lnadd(t0, xe, enc[l].n1g, enc[l].n1b, xe);
        do_ffn(xe, enc[l].W1, enc[l].b1, enc[l].W2, enc[l].b2, t0);
        lnadd(t0, xe, enc[l].n2g, enc[l].n2b, xe);
    }

    // ---- decoder ----
    hipMemcpyAsync(xd, d_in[1], MD * 4, hipMemcpyDeviceToDevice, stream);
    for (int l = 0; l < 4; ++l) {
        do_mha(xd, xd, dec[l].self, 1, t0);            // causal self-attn
        lnadd(t0, xd, dec[l].n1g, dec[l].n1b, xd);
        do_mha(xd, xe, dec[l].cross, 0, t0);           // cross-attn vs enc_out
        lnadd(t0, xd, dec[l].n2g, dec[l].n2b, xd);
        do_ffn(xd, dec[l].W1, dec[l].b1, dec[l].W2, dec[l].b2, t0);
        lnadd(t0, xd, dec[l].n3g, dec[l].n3b, xd);
    }

    hipMemcpyAsync(d_out, xd, MD * 4, hipMemcpyDeviceToDevice, stream);
}
